// VectorQuantizer_16441134809238
// MI455X (gfx1250) — matrix operand order repaired and checked by reading, not compiled, not run on hardware
//
#include <hip/hip_runtime.h>

// ---------------------------------------------------------------------------
// VQ-VAE vector quantizer for MI455X (gfx1250, wave32, WMMA).
//   x:         [16,4096,256] f32  -> M=65536 rows, E=256
//   embedding: [1024,256]    f32  -> N=1024 codes
// Outputs (concatenated in d_out): quantized [M,E] f32, loss f32, perplexity f32
//
// Strategy: split-bf16 GEMM (x_hi*e_hi + x_hi*e_lo + x_lo*e_hi) via
// v_wmma_f32_16x16x32_bf16 for near-f32 argmin accuracy at bf16 matrix
// throughput; fused per-row argmin (never materialize the 65536x1024 distance
// matrix); exact f32 loss from the gathered codebook rows.
// ---------------------------------------------------------------------------

typedef __attribute__((ext_vector_type(16))) __bf16 v16bf;
typedef __attribute__((ext_vector_type(8)))  float  v8f;

#define M_ROWS   65536
#define E_DIM    256
#define N_CODES  1024
#define MB       128            // rows per workgroup (8 waves x 16 rows)
#define NT_COUNT (N_CODES / 16) // 64 n-tiles
#define KT_COUNT (E_DIM / 32)   // 8 k-tiles
#define G2_BLOCKS ((M_ROWS * E_DIM / 4) / 256) // 16384

// ---------------- kernel 0: codebook prep (bf16 split + ||e||^2) -----------
__global__ __launch_bounds__(256) void vq_prep_kernel(
    const float* __restrict__ emb, __bf16* __restrict__ eh,
    __bf16* __restrict__ el, float* __restrict__ enorm,
    float* __restrict__ counts)
{
  __shared__ float red[256];
  const int n = blockIdx.x;
  const int t = threadIdx.x;
  const float e = emb[n * E_DIM + t];
  const __bf16 h = (__bf16)e;
  eh[n * E_DIM + t] = h;
  el[n * E_DIM + t] = (__bf16)(e - (float)h);
  red[t] = e * e;
  __syncthreads();
  for (int s = 128; s > 0; s >>= 1) {
    if (t < s) red[t] += red[t + s];
    __syncthreads();
  }
  if (t == 0) { enorm[n] = red[0]; counts[n] = 0.0f; }
}

// ---------------- kernel 1: fused split-bf16 GEMM + per-row argmin ---------
__global__ __launch_bounds__(256) void vq_argmin_kernel(
    const float* __restrict__ x, const __bf16* __restrict__ eh,
    const __bf16* __restrict__ el, const float* __restrict__ enorm,
    int* __restrict__ idx_out)
{
  const int lane = threadIdx.x & 31;
  const int wave = threadIdx.x >> 5;
  const int m0   = blockIdx.x * MB + wave * 16;
  const int row  = lane & 15;       // A row / B col / D col owned by this lane
  const int half = lane >> 4;       // lane half selects k sub-block

  // ---- Load this wave's 16 rows of x, split to bf16 hi/lo, keep in VGPRs.
  // A operand, contiguous k order used here (16x32, MxK): lanes 0-15 -> M=lane, k-base 0;
  // lanes 16-31 -> M=lane-16, k-base 16. Element e of v16: k = kbase + e,
  // pairs packed low-k-first per 32-bit VGPR.
  v16bf a_hi[KT_COUNT], a_lo[KT_COUNT];
  {
    const float* px = x + (size_t)(m0 + row) * E_DIM;
    const int abase = half * 16;
#pragma unroll
    for (int kt = 0; kt < KT_COUNT; ++kt) {
      const float* p = px + kt * 32 + abase;
      float4 f0 = *(const float4*)(p);
      float4 f1 = *(const float4*)(p + 4);
      float4 f2 = *(const float4*)(p + 8);
      float4 f3 = *(const float4*)(p + 12);
      float f[16] = {f0.x, f0.y, f0.z, f0.w, f1.x, f1.y, f1.z, f1.w,
                     f2.x, f2.y, f2.z, f2.w, f3.x, f3.y, f3.z, f3.w};
#pragma unroll
      for (int e = 0; e < 16; ++e) {
        const __bf16 h = (__bf16)f[e];
        a_hi[kt][e] = h;
        a_lo[kt][e] = (__bf16)(f[e] - (float)h);
      }
    }
  }

  float best_s[8];
  int   best_i[8];
#pragma unroll
  for (int v = 0; v < 8; ++v) { best_s[v] = 3.0e38f; best_i[v] = 0; }

  // B operand, contiguous k order used here (32x16, KxN): lane holds column n = lane&15; lanes 0-15
  // hold k=0..15, lanes 16-31 hold k=16..31 (16 consecutive bf16 per lane).
  const int bbase = half * 16;

  for (int nt = 0; nt < NT_COUNT; ++nt) {
    const int n = nt * 16 + row;
    const __bf16* ph = eh + (size_t)n * E_DIM + bbase;
    const __bf16* pl = el + (size_t)n * E_DIM + bbase;

    if (nt + 1 < NT_COUNT) {                 // prefetch next B tile (L2-hot)
      __builtin_prefetch(ph + 16 * E_DIM, 0, 3);
      __builtin_prefetch(pl + 16 * E_DIM, 0, 3);
    }

    // Three independent accumulator chains -> overlapped WMMA issue
    v8f acc_hh = {}, acc_hl = {}, acc_lh = {};
#pragma unroll
    for (int kt = 0; kt < KT_COUNT; ++kt) {
      const v16bf bh = *(const v16bf*)(ph + kt * 32);
      const v16bf bl = *(const v16bf*)(pl + kt * 32);
      acc_hh = __builtin_amdgcn_wmma_f32_16x16x32_bf16(
          false, a_hi[kt], false, bh, (short)0, acc_hh, false, false);
      acc_hl = __builtin_amdgcn_wmma_f32_16x16x32_bf16(
          false, a_hi[kt], false, bl, (short)0, acc_hl, false, false);
      acc_lh = __builtin_amdgcn_wmma_f32_16x16x32_bf16(
          false, a_lo[kt], false, bh, (short)0, acc_lh, false, false);
    }

    // score = ||e_n||^2 - 2 * x.e_n   (||x||^2 constant per row -> dropped)
    const float en = enorm[n];
#pragma unroll
    for (int v = 0; v < 8; ++v) {
      const float dot = acc_hh[v] + acc_hl[v] + acc_lh[v];
      const float s = en - 2.0f * dot;
      if (s < best_s[v]) { best_s[v] = s; best_i[v] = n; }
    }
  }

  // D layout: lane holds column n = lane&15, rows M = v + 8*half (v=0..7).
  // Reduce argmin across the 16 lanes of each half (masks 1..8 stay in-half).
#pragma unroll
  for (int m = 1; m < 16; m <<= 1) {
#pragma unroll
    for (int v = 0; v < 8; ++v) {
      const float s2 = __shfl_xor(best_s[v], m, 32);
      const int   i2 = __shfl_xor(best_i[v], m, 32);
      if (s2 < best_s[v] || (s2 == best_s[v] && i2 < best_i[v])) {
        best_s[v] = s2; best_i[v] = i2;
      }
    }
  }
  if (row == 0) {
#pragma unroll
    for (int v = 0; v < 8; ++v)
      idx_out[m0 + half * 8 + v] = best_i[v];
  }
}

// ---------------- kernel 2: gather + quantized out + loss partials + hist --
__global__ __launch_bounds__(256) void vq_gather_kernel(
    const float* __restrict__ x, const float* __restrict__ emb,
    const int* __restrict__ idx, float* __restrict__ out,
    float* __restrict__ partial, float* __restrict__ counts)
{
  __shared__ float red[256];
  const int gid  = blockIdx.x * 256 + threadIdx.x;
  const size_t base = (size_t)gid * 4;          // 4 f32 per thread, row-aligned
  const int m = (int)(base >> 8);               // E_DIM == 256
  const int c = (int)(base & 255);
  const int code = idx[m];

  const float4 q  = *(const float4*)(emb + (size_t)code * E_DIM + c);
  const float4 xv = *(const float4*)(x + base);
  *(float4*)(out + base) = q;                   // quantized_st value == quantized

  const float dx = q.x - xv.x, dy = q.y - xv.y;
  const float dz = q.z - xv.z, dw = q.w - xv.w;
  red[threadIdx.x] = dx * dx + dy * dy + dz * dz + dw * dw;

  if (c == 0) atomicAdd(&counts[code], 1.0f);   // one vote per row

  __syncthreads();
  for (int s = 128; s > 0; s >>= 1) {
    if (threadIdx.x < s) red[threadIdx.x] += red[threadIdx.x + s];
    __syncthreads();
  }
  if (threadIdx.x == 0) partial[blockIdx.x] = red[0];
}

// ---------------- kernel 3: finalize loss + perplexity ---------------------
__global__ __launch_bounds__(1024) void vq_final_kernel(
    const float* __restrict__ partial, const float* __restrict__ counts,
    float* __restrict__ out_loss, float* __restrict__ out_perp)
{
  __shared__ float rs[1024];
  __shared__ float rh[1024];
  const int t = threadIdx.x;
  float s = 0.0f;
  for (int i = t; i < G2_BLOCKS; i += 1024) s += partial[i];
  const float p = counts[t] * (1.0f / (float)M_ROWS);
  rs[t] = s;
  rh[t] = p * logf(p + 1e-10f);
  __syncthreads();
  for (int st = 512; st > 0; st >>= 1) {
    if (t < st) { rs[t] += rs[t + st]; rh[t] += rh[t + st]; }
    __syncthreads();
  }
  if (t == 0) {
    const float mse = rs[0] / ((float)M_ROWS * (float)E_DIM);
    *out_loss = 1.25f * mse;        // q_latent + 0.25 * e_latent, both == mse
    *out_perp = expf(-rh[0]);
  }
}

// ---------------------------------------------------------------------------
extern "C" void kernel_launch(void* const* d_in, const int* in_sizes, int n_in,
                              void* d_out, int out_size, void* d_ws, size_t ws_size,
                              hipStream_t stream) {
  const float* x   = (const float*)d_in[0];   // [16,4096,256] f32
  const float* emb = (const float*)d_in[1];   // [1024,256] f32
  float* out = (float*)d_out;

  // workspace layout
  char* ws = (char*)d_ws;
  __bf16* eh     = (__bf16*)(ws);                              // 512 KB
  __bf16* el     = (__bf16*)(ws + 512 * 1024);                 // 512 KB
  float*  enorm  = (float*) (ws + 1024 * 1024);                // 4 KB
  float*  counts = (float*) (ws + 1024 * 1024 + 4096);         // 4 KB
  int*    idx    = (int*)   (ws + 1024 * 1024 + 8192);         // 256 KB
  float*  part   = (float*) (ws + 1024 * 1024 + 8192 + 262144);// 64 KB

  vq_prep_kernel  <<<N_CODES, 256, 0, stream>>>(emb, eh, el, enorm, counts);
  vq_argmin_kernel<<<M_ROWS / MB, 256, 0, stream>>>(x, eh, el, enorm, idx);
  vq_gather_kernel<<<G2_BLOCKS, 256, 0, stream>>>(x, emb, idx, out, part, counts);
  vq_final_kernel <<<1, 1024, 0, stream>>>(part, counts,
                                           out + (size_t)M_ROWS * E_DIM,
                                           out + (size_t)M_ROWS * E_DIM + 1);
}
